// MBAM_9534827397173
// MI455X (gfx1250) — matrix operand order repaired and checked by reading, not compiled, not run on hardware
//
#include <hip/hip_runtime.h>

#define BDIM 8
#define CCH 128
#define LL 4096
#define BL 32768   // BDIM * LL
#define NCH 64     // chunks per sequence
#define CLEN 64    // chunk length (NCH*CLEN == LL)

typedef __attribute__((ext_vector_type(16))) __bf16 v16bf;
typedef __attribute__((ext_vector_type(8)))  float  v8f;

#define DEV static __device__ __forceinline__

DEV float sigf(float x) { return 1.0f / (1.0f + __expf(-x)); }
DEV float siluf(float x) { return x * sigf(x); }

DEV __bf16 f2bf(float f) {
    unsigned int u = __float_as_uint(f);
    u += 0x7FFFu + ((u >> 16) & 1u);           // round-to-nearest-even
    unsigned short h = (unsigned short)(u >> 16);
    return __builtin_bit_cast(__bf16, h);
}
DEV float bf2f(__bf16 b) {
    unsigned int u = ((unsigned int)__builtin_bit_cast(unsigned short, b)) << 16;
    return __uint_as_float(u);
}

// ---------------------------------------------------------------------------
// bf16 WMMA GEMM: C[M,N] = A[M,K] * Bt[N,K]^T (+bias[n]).
// One wave computes a 16x(16*NT) strip: A fragment loaded once per K-step and
// reused across NT back-to-back v_wmma issues. NT is a compile-time constant
// so there is NO control flow around the WMMAs (EXEC stays all-ones, acc stays
// in fixed registers). Requires tilesN % NT == 0.
// ---------------------------------------------------------------------------
template<int NT>
__global__ __launch_bounds__(128) void k_gemm(
    const __bf16* __restrict__ A, const __bf16* __restrict__ Bt,
    float* __restrict__ C, const float* __restrict__ bias,
    int M, int N, int K)
{
    union Frag { v16bf v; unsigned int u[8]; };
    int lane = threadIdx.x & 31;
    int wave = threadIdx.x >> 5;
    int tilesN = N >> 4;
    int tilesNB = tilesN / NT;                  // exact by construction
    int groups = (M >> 4) * tilesNB;
    int gid = blockIdx.x * (blockDim.x >> 5) + wave;
    if (gid >= groups) return;                  // wave-uniform; EXEC stays all-1
    int tm = gid / tilesNB;
    int tn0 = (gid - tm * tilesNB) * NT;
    int mr = lane & 15, half = lane >> 4;

    const __bf16* pA  = A  + (size_t)(tm * 16 + mr) * K;
    const __bf16* pB0 = Bt + (size_t)(tn0 * 16 + mr) * K;

    v8f acc[NT] = {};
    for (int k0 = 0; k0 < K; k0 += 32) {
        Frag a;
        #pragma unroll
        for (int j = 0; j < 8; ++j) {
            int ka = k0 + ((j < 4) ? (2 * j) : (16 + 2 * (j - 4))) + 8 * half;
            a.u[j] = *(const unsigned int*)(pA + ka);
        }
        #pragma unroll
        for (int jt = 0; jt < NT; ++jt) {
            Frag b;
            const __bf16* pB = pB0 + (size_t)jt * 16 * K;
            #pragma unroll
            for (int j = 0; j < 8; ++j) {
                int kb = k0 + ((j < 4) ? (2 * j) : (16 + 2 * (j - 4))) + 8 * half;
                b.u[j] = *(const unsigned int*)(pB + kb);
            }
            acc[jt] = __builtin_amdgcn_wmma_f32_16x16x32_bf16(
                false, a.v, false, b.v, (short)0, acc[jt], false, false);
        }
    }
    #pragma unroll
    for (int jt = 0; jt < NT; ++jt) {
        int n = (tn0 + jt) * 16 + mr;
        float bv = bias ? bias[n] : 0.0f;
        #pragma unroll
        for (int r = 0; r < 8; ++r) {
            int m = tm * 16 + r + 8 * half;     // D layout: vgpr r -> row r+8*half
            C[(size_t)m * N + n] = acc[jt][r] + bv;
        }
    }
}

// f32 (R,C) row-major -> bf16 (C,R) row-major (weight transpose-convert)
__global__ void k_wt(const float* __restrict__ in, __bf16* __restrict__ out, int R, int Cc)
{
    int i = blockIdx.x * blockDim.x + threadIdx.x;
    if (i >= R * Cc) return;
    int r = i / Cc, c = i - r * Cc;
    out[(size_t)c * R + r] = f2bf(in[i]);
}

// x (B,C,L) f32 -> token-major bf16 (B*L, 128)
__global__ void k_xt(const float* __restrict__ x, __bf16* __restrict__ out)
{
    int i = blockIdx.x * blockDim.x + threadIdx.x;   // BL*128
    int c = i & 127; int t = i >> 7;
    int b = t >> 12; int l = t & 4095;
    out[i] = f2bf(x[(((size_t)(b * 128 + c)) << 12) + l]);
}

// depthwise 3x3 SAME + bias + SiLU : F1 f32 (BL,128) -> bf16 (BL,128)
__global__ void k_dw(const float* __restrict__ in, const float* __restrict__ w,
                     const float* __restrict__ bias, __bf16* __restrict__ out)
{
    int i = blockIdx.x * blockDim.x + threadIdx.x;   // BL*128
    int c = i & 127; int t = i >> 7;
    int b = t >> 12; int l = t & 4095;
    int hh = l >> 6, ww = l & 63;
    float acc = bias[c];
    const float* wk = w + c * 9;
    size_t tb = (size_t)b << 12;
    #pragma unroll
    for (int dy = -1; dy <= 1; ++dy) {
        int y = hh + dy; if (y < 0 || y > 63) continue;
        #pragma unroll
        for (int dx = -1; dx <= 1; ++dx) {
            int xw = ww + dx; if (xw < 0 || xw > 63) continue;
            acc += in[(tb + (size_t)(y * 64 + xw)) * 128 + c] * wk[(dy + 1) * 3 + (dx + 1)];
        }
    }
    out[i] = f2bf(siluf(acc));
}

// s f32 (BL,32) -> bf16 forward + bf16 sequence-flipped
__global__ void k_flips(const float* __restrict__ s, __bf16* __restrict__ sf,
                        __bf16* __restrict__ sr)
{
    int i = blockIdx.x * blockDim.x + threadIdx.x;   // BL*32
    int c = i & 31; int t = i >> 5;
    int b = t >> 12; int l = t & 4095;
    float v = s[i];
    __bf16 bv = f2bf(v);
    sf[i] = bv;
    sr[((((size_t)b) << 12) + (4095 - l)) * 32 + c] = bv;
}

// dt = softplus(s . w_dt + bias); dA = exp(-exp(A_log)*dt)
__global__ void k_dt(const __bf16* __restrict__ s, const float* __restrict__ win,
                     const float* __restrict__ dtb, const float* __restrict__ alog,
                     float* __restrict__ dtp, float* __restrict__ dab)
{
    int t = blockIdx.x * blockDim.x + threadIdx.x;
    if (t >= BL) return;
    float acc = dtb[0];
    const __bf16* sp = s + (size_t)t * 32;
    #pragma unroll
    for (int k = 0; k < 32; ++k) acc += bf2f(sp[k]) * win[k * 257 + 256];
    float spv = (acc > 20.0f) ? acc : log1pf(__expf(acc));
    dtp[t] = spv;
    dab[t] = __expf(-__expf(alog[0]) * spv);
}

// causal depthwise conv1d (k=4) + bias + SiLU over zx cols 64..255 -> F3 (BL,192)
__global__ void k_cconv(const float* __restrict__ zx, const float* __restrict__ w,
                        const float* __restrict__ bias, float* __restrict__ out)
{
    int i = blockIdx.x * blockDim.x + threadIdx.x;   // BL*192
    int c = i % 192; int t = i / 192;
    int l = t & 4095;
    float acc = bias[c];
    const float* wk = w + c * 4;
    const float* base = zx + (size_t)t * 256 + 64 + c;
    #pragma unroll
    for (int j = 0; j < 4; ++j) {
        int ls = l - 3 + j;
        if (ls >= 0) acc += base[(ptrdiff_t)(ls - l) * 256] * wk[j];
    }
    out[(size_t)t * 192 + c] = siluf(acc);
}

// ---------------------------------------------------------------------------
// Chunked linear scan (Mamba2 SSD style), 64 chunks x 64 steps per sequence.
// Thread layout (256/block): p = tid>>2 (head dim), q = tid&3 -> 16 n-values.
// ---------------------------------------------------------------------------

// pass 1: chunk-local end state S_c (h0 = 0) and chunk decay P_c = prod(dA)
__global__ __launch_bounds__(256) void k_scan1(
    const float* __restrict__ xbc, const float* __restrict__ dtp,
    const float* __restrict__ dab, float* __restrict__ S, float* __restrict__ P)
{
    int bc = blockIdx.x;                   // b*NCH + chunk
    int tid = threadIdx.x;
    int p = tid >> 2, q = tid & 3;
    float h[16];
    #pragma unroll
    for (int i = 0; i < 16; ++i) h[i] = 0.0f;
    float prod = 1.0f;
    size_t base = (size_t)bc * CLEN;
    for (int l = 0; l < CLEN; ++l) {
        size_t t = base + l;
        const float* row = xbc + t * 192;
        float dA = dab[t], dtv = dtp[t];
        float xsp = row[p];
        const float4* Bp = (const float4*)(row + 64 + q * 16);
        float4 Bq[4];
        #pragma unroll
        for (int j = 0; j < 4; ++j) Bq[j] = Bp[j];
        const float* Bv = (const float*)Bq;
        float dtx = dtv * xsp;
        prod *= dA;
        #pragma unroll
        for (int i = 0; i < 16; ++i) h[i] = dA * h[i] + dtx * Bv[i];
    }
    float* Sp = S + (size_t)bc * 4096 + p * 64 + q * 16;
    #pragma unroll
    for (int i = 0; i < 16; ++i) Sp[i] = h[i];
    if (tid == 0) P[bc] = prod;
}

// pass 2: inter-chunk scan, elementwise over the 64x64 state
__global__ void k_scan2(const float* __restrict__ S, const float* __restrict__ P,
                        float* __restrict__ Hs)
{
    int i = blockIdx.x * blockDim.x + threadIdx.x;   // B*4096
    int b = i >> 12; int e = i & 4095;
    float H = 0.0f;
    for (int c = 0; c < NCH; ++c) {
        size_t idx = (((size_t)b << 6) + c) * 4096 + e;
        Hs[idx] = H;                                  // state at chunk start
        H = P[(b << 6) + c] * H + S[idx];
    }
}

// pass 3: replay each chunk from its exact start state; emit y = h.C + D*xs
__global__ __launch_bounds__(256) void k_scan3(
    const float* __restrict__ xbc, const float* __restrict__ dtp,
    const float* __restrict__ dab, const float* __restrict__ Hs,
    const float* __restrict__ Dptr, float* __restrict__ yout)
{
    int bc = blockIdx.x;
    int tid = threadIdx.x;
    int p = tid >> 2, q = tid & 3;
    float Dv = Dptr[0];
    float h[16];
    const float* Hp = Hs + (size_t)bc * 4096 + p * 64 + q * 16;
    #pragma unroll
    for (int i = 0; i < 16; ++i) h[i] = Hp[i];
    size_t base = (size_t)bc * CLEN;
    for (int l = 0; l < CLEN; ++l) {
        size_t t = base + l;
        const float* row = xbc + t * 192;
        float dA = dab[t], dtv = dtp[t];
        float xsp = row[p];
        const float4* Bp = (const float4*)(row + 64 + q * 16);
        const float4* Cp = (const float4*)(row + 128 + q * 16);
        float4 Bq[4], Cq[4];
        #pragma unroll
        for (int j = 0; j < 4; ++j) { Bq[j] = Bp[j]; Cq[j] = Cp[j]; }
        const float* Bv = (const float*)Bq;
        const float* Cv = (const float*)Cq;
        float dtx = dtv * xsp;
        float acc = 0.0f;
        #pragma unroll
        for (int i = 0; i < 16; ++i) {
            h[i] = dA * h[i] + dtx * Bv[i];
            acc = fmaf(h[i], Cv[i], acc);
        }
        acc += __shfl_xor(acc, 1, 32);
        acc += __shfl_xor(acc, 2, 32);
        if (q == 0) yout[t * 64 + p] = acc + Dv * xsp;
    }
}

// y *= silu(z); RMS-norm over 64; * norm_w; -> bf16
__global__ __launch_bounds__(64) void k_gn(
    const float* __restrict__ y, const float* __restrict__ zx,
    const float* __restrict__ nw, __bf16* __restrict__ out)
{
    int t = blockIdx.x; int p = threadIdx.x;   // 64 threads = 2 waves
    float yv = y[(size_t)t * 64 + p];
    float z = zx[(size_t)t * 256 + p];
    yv *= siluf(z);
    float sq = yv * yv;
    #pragma unroll
    for (int m = 1; m < 32; m <<= 1) sq += __shfl_xor(sq, m, 32);
    __shared__ float red[2];
    if ((p & 31) == 0) red[p >> 5] = sq;
    __syncthreads();
    float tot = red[0] + red[1];
    out[(size_t)t * 64 + p] = f2bf(yv * rsqrtf(tot * (1.0f / 64.0f) + 1e-5f) * nw[p]);
}

// ysum = ydm0[t] + ydm1[flip(t)]  -> bf16
__global__ void k_comb(const float* __restrict__ a, const float* __restrict__ bf,
                       __bf16* __restrict__ out)
{
    int i = blockIdx.x * blockDim.x + threadIdx.x;   // BL*32
    int c = i & 31; int t = i >> 5;
    int b = t >> 12; int l = t & 4095;
    float v = a[i] + bf[((((size_t)b) << 12) + (4095 - l)) * 32 + c];
    out[i] = f2bf(v);
}

// deterministic BN stats: per-block partials (512 blocks x 64 rows)
__global__ __launch_bounds__(256) void k_bnstat(const float* __restrict__ o,
                                                float* __restrict__ part)
{
    __shared__ float ls[256], ls2[256];
    int tid = threadIdx.x; int c = tid & 127; int g = tid >> 7;
    int rend = blockIdx.x * 64 + 64;
    float s = 0.0f, s2 = 0.0f;
    for (int r = blockIdx.x * 64 + g; r < rend; r += 2) {
        float v = o[(size_t)r * 128 + c];
        s += v; s2 += v * v;
    }
    ls[tid] = s; ls2[tid] = s2;
    __syncthreads();
    if (g == 0) {
        part[(size_t)blockIdx.x * 256 + c]       = s  + ls[tid + 128];
        part[(size_t)blockIdx.x * 256 + 128 + c] = s2 + ls2[tid + 128];
    }
}

__global__ __launch_bounds__(256) void k_bnred(const float* __restrict__ part,
                                               float* __restrict__ bn)
{
    int j = threadIdx.x;  // 256
    float acc = 0.0f;
    for (int b = 0; b < 512; ++b) acc += part[(size_t)b * 256 + j];
    bn[j] = acc;
}

__global__ void k_bnapply(const float* __restrict__ o, const float* __restrict__ bn,
                          const float* __restrict__ g, const float* __restrict__ bt,
                          __bf16* __restrict__ out)
{
    int i = blockIdx.x * blockDim.x + threadIdx.x;   // BL*128
    int c = i & 127;
    float mean = bn[c] * (1.0f / 32768.0f);
    float var = bn[128 + c] * (1.0f / 32768.0f) - mean * mean;
    out[i] = f2bf((o[i] - mean) * rsqrtf(var + 1e-5f) * g[c] + bt[c]);
}

// out[b,c,l] = x * (1 + sigmoid(gpre + b4))   (channel-major output)
__global__ void k_final(const float* __restrict__ gpre, const float* __restrict__ c4b,
                        const float* __restrict__ x, float* __restrict__ out)
{
    int i = blockIdx.x * blockDim.x + threadIdx.x;   // B*C*L, l fastest
    int l = i & 4095; int bc = i >> 12;
    int c = bc & 127; int b = bc >> 7;
    size_t t = ((size_t)b << 12) + l;
    float gv = sigf(gpre[t * 128 + c] + c4b[c]);
    float xv = x[i];
    out[i] = gv * xv + xv;
}

extern "C" void kernel_launch(void* const* d_in, const int* in_sizes, int n_in,
                              void* d_out, int out_size, void* d_ws, size_t ws_size,
                              hipStream_t stream)
{
    const float* x        = (const float*)d_in[0];
    const float* lin1_w   = (const float*)d_in[1];
    const float* lin1_b   = (const float*)d_in[2];
    const float* dw_w     = (const float*)d_in[3];
    const float* dw_b     = (const float*)d_in[4];
    const float* fc_in_w  = (const float*)d_in[5];
    const float* mam_in_w = (const float*)d_in[6];
    const float* mam_cw   = (const float*)d_in[7];
    const float* mam_cb   = (const float*)d_in[8];
    const float* mam_dtb  = (const float*)d_in[9];
    const float* mam_Alog = (const float*)d_in[10];
    const float* mam_D    = (const float*)d_in[11];
    const float* mam_nw   = (const float*)d_in[12];
    const float* mam_ow   = (const float*)d_in[13];
    const float* fc_out_w = (const float*)d_in[14];
    const float* bn_g     = (const float*)d_in[15];
    const float* bn_b     = (const float*)d_in[16];
    const float* conv4_w  = (const float*)d_in[17];
    const float* conv4_b  = (const float*)d_in[18];
    float* out = (float*)d_out;

    char* ws = (char*)d_ws;
    size_t off = 0;
    auto take = [&](size_t bytes) -> void* {
        void* p = ws + off;
        off += (bytes + 255) & ~(size_t)255;
        return p;
    };

    __bf16* w1t    = (__bf16*)take((size_t)128 * 128 * 2);
    __bf16* fcint  = (__bf16*)take((size_t)32 * 128 * 2);
    __bf16* mint0  = (__bf16*)take((size_t)257 * 32 * 2);
    __bf16* mint1  = (__bf16*)take((size_t)257 * 32 * 2);
    __bf16* moutt0 = (__bf16*)take((size_t)32 * 64 * 2);
    __bf16* moutt1 = (__bf16*)take((size_t)32 * 64 * 2);
    __bf16* fcoutt = (__bf16*)take((size_t)128 * 32 * 2);
    __bf16* c4t    = (__bf16*)take((size_t)128 * 128 * 2);

    __bf16* A1  = (__bf16*)take((size_t)BL * 128 * 2);  // xt / o2 / o3b
    float*  F1  = (float*)take((size_t)BL * 128 * 4);   // o1 / (S|Hs) / o3
    float*  Fs  = (float*)take((size_t)BL * 32 * 4);    // s f32
    __bf16* A3  = (__bf16*)take((size_t)BL * 32 * 2);   // s bf16
    __bf16* A4  = (__bf16*)take((size_t)BL * 32 * 2);   // s flipped
    float*  F2  = (float*)take((size_t)BL * 256 * 4);   // zxbcdt / gpre
    float*  F3  = (float*)take((size_t)BL * 192 * 4);   // conv'd xBC
    float*  dtp = (float*)take((size_t)BL * 4);
    float*  dab = (float*)take((size_t)BL * 4);
    float*  F4  = (float*)take((size_t)BL * 64 * 4);    // scan y
    __bf16* A5  = (__bf16*)take((size_t)BL * 64 * 2);   // gated/normed y
    float*  F5a = (float*)take((size_t)BL * 32 * 4);
    float*  F5b = (float*)take((size_t)BL * 32 * 4);
    __bf16* A6  = (__bf16*)take((size_t)BL * 32 * 2);
    float*  bnp = (float*)take((size_t)512 * 256 * 4);
    float*  bn  = (float*)take((size_t)256 * 4);
    float*  Pbuf = (float*)take((size_t)BDIM * NCH * 4);

    // aliases (regions dead at time of reuse)
    float*  Sbuf = F1;                                   // 8 MB (dir loop only)
    float*  Hbuf = F1 + (size_t)BDIM * NCH * 4096;       // 8 MB (dir loop only)
    float*  F6 = F1;   // fc_out result (o1/S/H dead)
    __bf16* A7 = A1;   // bn-applied bf16 (o2 dead)
    float*  F7 = F2;   // conv4 pre-sigmoid (zx dead)

    auto gemm = [&](const __bf16* A, const __bf16* Bt, float* C, const float* bias,
                    int M, int N, int K) {
        int tilesN = N / 16;
        if (tilesN % 4 == 0) {
            int groups = (M / 16) * (tilesN / 4);
            k_gemm<4><<<dim3((groups + 3) / 4), dim3(128), 0, stream>>>(
                A, Bt, C, bias, M, N, K);
        } else {
            int groups = (M / 16) * (tilesN / 2);
            k_gemm<2><<<dim3((groups + 3) / 4), dim3(128), 0, stream>>>(
                A, Bt, C, bias, M, N, K);
        }
    };
    auto gcd1 = [](int n) { return dim3((n + 255) / 256); };

    // --- weight transposes (tiny) ---
    k_wt<<<gcd1(128 * 128), dim3(256), 0, stream>>>(lin1_w, w1t, 128, 128);
    k_wt<<<gcd1(128 * 32), dim3(256), 0, stream>>>(fc_in_w, fcint, 128, 32);
    k_wt<<<gcd1(32 * 257), dim3(256), 0, stream>>>(mam_in_w, mint0, 32, 257);
    k_wt<<<gcd1(32 * 257), dim3(256), 0, stream>>>(mam_in_w + 32 * 257, mint1, 32, 257);
    k_wt<<<gcd1(64 * 32), dim3(256), 0, stream>>>(mam_ow, moutt0, 64, 32);
    k_wt<<<gcd1(64 * 32), dim3(256), 0, stream>>>(mam_ow + 64 * 32, moutt1, 64, 32);
    k_wt<<<gcd1(32 * 128), dim3(256), 0, stream>>>(fc_out_w, fcoutt, 32, 128);
    k_wt<<<gcd1(128 * 128), dim3(256), 0, stream>>>(conv4_w, c4t, 128, 128);

    // --- stage A: x -> token-major bf16; lin1 GEMM; dw 3x3 + SiLU ---
    k_xt<<<gcd1(BL * 128), dim3(256), 0, stream>>>(x, A1);
    gemm(A1, w1t, F1, lin1_b, BL, 128, 128);
    k_dw<<<gcd1(BL * 128), dim3(256), 0, stream>>>(F1, dw_w, dw_b, A1);

    // --- stage B: fc_in GEMM; build forward + flipped s (bf16) ---
    gemm(A1, fcint, Fs, nullptr, BL, 32, 128);
    k_flips<<<gcd1(BL * 32), dim3(256), 0, stream>>>(Fs, A3, A4);

    // --- per-direction Mamba2 ---
    for (int d = 0; d < 2; ++d) {
        const __bf16* sd = d ? A4 : A3;
        const __bf16* mint = d ? mint1 : mint0;
        const __bf16* moutt = d ? moutt1 : moutt0;
        gemm(sd, mint, F2, nullptr, BL, 256, 32);                 // z|xBC (dt col sep.)
        k_dt<<<gcd1(BL), dim3(256), 0, stream>>>(sd, mam_in_w + d * 32 * 257,
                                                 mam_dtb + d, mam_Alog + d, dtp, dab);
        k_cconv<<<gcd1(BL * 192), dim3(256), 0, stream>>>(F2, mam_cw + d * 192 * 4,
                                                          mam_cb + d * 192, F3);
        // chunked scan: 512-way parallel
        k_scan1<<<dim3(BDIM * NCH), dim3(256), 0, stream>>>(F3, dtp, dab, Sbuf, Pbuf);
        k_scan2<<<gcd1(BDIM * 4096), dim3(256), 0, stream>>>(Sbuf, Pbuf, Hbuf);
        k_scan3<<<dim3(BDIM * NCH), dim3(256), 0, stream>>>(F3, dtp, dab, Hbuf,
                                                            mam_D + d, F4);
        k_gn<<<dim3(BL), dim3(64), 0, stream>>>(F4, F2, mam_nw + d * 64, A5);
        gemm(A5, moutt, d ? F5b : F5a, nullptr, BL, 32, 64);
    }

    // --- combine directions; fc_out; BN; conv4; gate residual ---
    k_comb<<<gcd1(BL * 32), dim3(256), 0, stream>>>(F5a, F5b, A6);
    gemm(A6, fcoutt, F6, nullptr, BL, 128, 32);
    k_bnstat<<<dim3(512), dim3(256), 0, stream>>>(F6, bnp);
    k_bnred<<<dim3(1), dim3(256), 0, stream>>>(bnp, bn);
    k_bnapply<<<gcd1(BL * 128), dim3(256), 0, stream>>>(F6, bn, bn_g, bn_b, A7);
    gemm(A7, c4t, F7, nullptr, BL, 128, 128);
    k_final<<<gcd1(BDIM * CCH * LL), dim3(256), 0, stream>>>(F7, conv4_b, x, out);

    (void)in_sizes; (void)n_in; (void)out_size; (void)ws_size;
}
